// MemoryController_35270271435167
// MI455X (gfx1250) — matrix operand order repaired and checked by reading, not compiled, not run on hardware
//
#include <hip/hip_runtime.h>
#include <hip/hip_bf16.h>

// ---------------------------------------------------------------------------
// MI455X (gfx1250) implementation.
// Heavy FFN GEMMs -> v_wmma_f32_16x16x32_bf16 (wave32, 16x16x32 tiles),
// A-tile staging via Tensor Data Mover (tensor_load_to_lds + s_wait_tensorcnt),
// hardware LDS padding via D#.pad_enable.
// Everything else is HBM-bandwidth bound -> fp32 streaming kernels.
// ---------------------------------------------------------------------------

typedef __attribute__((ext_vector_type(16))) __bf16 v16bf;
typedef __attribute__((ext_vector_type(8)))  float  v8f;
typedef __attribute__((ext_vector_type(4)))  __bf16 v4bf;
typedef __attribute__((ext_vector_type(4)))  unsigned int v4u;
typedef __attribute__((ext_vector_type(8)))  int v8i;
typedef __attribute__((ext_vector_type(4)))  int v4i;

#define LN_EPS 1e-5f

// ---------------------------------------------------------------------------
// fp32 -> bf16 bulk convert (weights)
// ---------------------------------------------------------------------------
__global__ __launch_bounds__(256) void f32_to_bf16_kernel(
    const float* __restrict__ src, __bf16* __restrict__ dst, size_t n4) {
  size_t i = (size_t)blockIdx.x * 256 + threadIdx.x;
  if (i >= n4) return;
  float4 v = reinterpret_cast<const float4*>(src)[i];
  v4bf o;
  o[0] = (__bf16)v.x; o[1] = (__bf16)v.y; o[2] = (__bf16)v.z; o[3] = (__bf16)v.w;
  reinterpret_cast<v4bf*>(dst)[i] = o;
}

// ---------------------------------------------------------------------------
// h = embed[seq]  (gather, fp32 -> bf16)
// ---------------------------------------------------------------------------
__global__ __launch_bounds__(256) void gather_embed_bf16_kernel(
    const int* __restrict__ seq, const float* __restrict__ embed,
    __bf16* __restrict__ h, int D) {
  size_t idx = (size_t)blockIdx.x * 256 + threadIdx.x;   // one float4 per thread
  int dq = D >> 2;
  size_t row = idx / dq;
  int off = (int)(idx % dq);
  const float4 v =
      reinterpret_cast<const float4*>(embed + (size_t)seq[row] * D)[off];
  v4bf o;
  o[0] = (__bf16)v.x; o[1] = (__bf16)v.y; o[2] = (__bf16)v.z; o[3] = (__bf16)v.w;
  reinterpret_cast<v4bf*>(h)[idx] = o;
}

// ---------------------------------------------------------------------------
// TDM: 2-D tile load Global -> LDS with hardware row padding.
// tile = tile_h rows x tile_w_dw DWORDs, global row stride row_stride_dw DWORDs.
// pad codes (ISA D# group1): interval c -> 2^(c+1) DWORDs, amount c -> c+1 DWORDs.
// Uses the 6-arg clang-23 builtin: (v4u, v8i, v4i, v4i, v8i, i32 cpol).
// ---------------------------------------------------------------------------
__device__ __forceinline__ void tdm_load_2d(unsigned lds_addr, const void* gptr,
                                            unsigned tile_w_dw, unsigned tile_h,
                                            unsigned long long row_stride_dw,
                                            unsigned pad_interval_code,
                                            unsigned pad_amount_code) {
  unsigned long long ga = (unsigned long long)(uintptr_t)gptr;
  v4u g0;
  g0[0] = 1u;                                             // count=1, user D#
  g0[1] = lds_addr;                                       // LDS byte address
  g0[2] = (unsigned)(ga & 0xFFFFFFFFu);                   // global_addr[31:0]
  g0[3] = (unsigned)((ga >> 32) & 0x01FFFFFFu) | (2u << 30); // addr[56:32]|type=2
  v8i g1;
  unsigned dw0 = (2u << 16);                              // data_size = 4B
  dw0 |= (1u << 20) | (pad_interval_code << 22) | (pad_amount_code << 25);
  g1[0] = (int)dw0;
  g1[1] = (int)((tile_w_dw & 0xFFFFu) << 16);             // tensor_dim0 lo16
  g1[2] = (int)(((tile_w_dw >> 16) & 0xFFFFu) |           // tensor_dim0 hi16
                ((tile_h & 0xFFFFu) << 16));              // tensor_dim1 lo16
  g1[3] = (int)(((tile_h >> 16) & 0xFFFFu) |              // tensor_dim1 hi16
                ((tile_w_dw & 0xFFFFu) << 16));           // tile_dim0
  g1[4] = (int)(tile_h & 0xFFFFu);                        // tile_dim1; tile_dim2=0
  g1[5] = (int)(row_stride_dw & 0xFFFFFFFFull);           // stride0 lo32
  g1[6] = (int)((row_stride_dw >> 32) & 0xFFFFull);       // stride0 hi16
  g1[7] = 0;
  v4i z4 = {0, 0, 0, 0};                                  // groups 2/3 unused (2-D)
  v8i z8 = {0, 0, 0, 0, 0, 0, 0, 0};
  __builtin_amdgcn_tensor_load_to_lds(g0, g1, z4, z4, z8, 0);
}

// ---------------------------------------------------------------------------
// WMMA GEMM: C[M,N] = A[M,K](bf16,row) @ B[K,N](bf16,row) + bias[N]
// epi==0: relu, store bf16.   epi==1: store fp32.
// Block tile 128x128, 8 waves, each wave: 16 rows x 128 cols (8 WMMA tiles).
// A tile: TDM DMA (wave 0) with HW padding.  B tile: cooperative transpose.
// ---------------------------------------------------------------------------
#define LDSTR 40   // padded k-stride (bf16 elems): 16 DW data + 4 DW pad

__global__ __launch_bounds__(256) void wmma_ffn_gemm(
    const __bf16* __restrict__ A, const __bf16* __restrict__ Bw,
    const float* __restrict__ bias, void* __restrict__ Cout,
    int N, int K, int epi) {
  __shared__ __bf16 As[2][128 * LDSTR];
  __shared__ __bf16 Bs[2][128 * LDSTR];

  const int tid  = threadIdx.x;
  const int wave = tid >> 5;
  const int lane = tid & 31;
  const int hi   = lane >> 4;      // half-wave select
  const int l16  = lane & 15;
  const int m0   = blockIdx.y * 128;
  const int n0   = blockIdx.x * 128;

  v8f acc[8];
#pragma unroll
  for (int j = 0; j < 8; ++j)
    acc[j] = (v8f){0.f, 0.f, 0.f, 0.f, 0.f, 0.f, 0.f, 0.f};

  const int bkrow = tid >> 3;          // 0..31    (B stage: 16 bf16/thread)
  const int bncol = (tid & 7) * 16;    // 0..112

  // A tile via Tensor Data Mover: 128 rows x 16 DW, pad 4 DW / 16 DW -> LDSTR.
  auto stage_a = [&](int buf, int k0) {
    if (wave == 0)
      tdm_load_2d((unsigned)(uintptr_t)&As[buf][0],
                  A + (size_t)m0 * K + k0,
                  /*tile_w_dw=*/16, /*tile_h=*/128,
                  /*row_stride_dw=*/(unsigned long long)(K >> 1),
                  /*pad_interval(16DW)=*/3u, /*pad_amount(4DW)=*/3u);
  };
  // B tile rows k0..k0+31, cols n0..n0+127 -> transposed into Bs[n][k]
  auto stage_b = [&](int buf, int k0) {
    union { float4 f[2]; __bf16 e[16]; } bb;
    const float4* gb = reinterpret_cast<const float4*>(
        Bw + (size_t)(k0 + bkrow) * N + n0 + bncol);
    bb.f[0] = gb[0]; bb.f[1] = gb[1];
#pragma unroll
    for (int i = 0; i < 16; ++i)
      Bs[buf][(bncol + i) * LDSTR + bkrow] = bb.e[i];
    if (k0 + 64 < K)   // prefetch two k-steps ahead (global_prefetch_b8)
      __builtin_prefetch(Bw + (size_t)(k0 + 64 + bkrow) * N + n0 + bncol, 0, 1);
  };

  stage_a(0, 0);
  stage_b(0, 0);
  int buf = 0;
  for (int k0 = 0; k0 < K; k0 += 32) {
    if (wave == 0) __builtin_amdgcn_s_wait_tensorcnt(0);  // A tile landed
    __syncthreads();                                      // publish As/Bs[buf]
    if (k0 + 32 < K) {
      stage_a(buf ^ 1, k0 + 32);
      stage_b(buf ^ 1, k0 + 32);
    }

    // A fragment: lane holds row (wave*16 + l16), K = {kb..kb+7, kb+16..kb+23},
    // kb = hi*8  (ISA 16-bit A 16x32 layout)
    union { v16bf v; float4 f[2]; } afrag;
    const float4* ap = reinterpret_cast<const float4*>(
        &As[buf][(wave * 16 + l16) * LDSTR + hi * 8]);
    afrag.f[0] = ap[0];
    afrag.f[1] = ap[2];   // +16 elements = +32B

    // Preload ALL B fragments, then back-to-back WMMAs (single dscnt drain).
    union { v16bf v; float4 f[2]; } bfrag[8];
#pragma unroll
    for (int j = 0; j < 8; ++j) {
      // lane holds column (j*16 + l16), K = {hi*8..hi*8+7, 16+hi*8..16+hi*8+7}
      const float4* bp = reinterpret_cast<const float4*>(
          &Bs[buf][(j * 16 + l16) * LDSTR + hi * 8]);
      bfrag[j].f[0] = bp[0];
      bfrag[j].f[1] = bp[2];
    }
#pragma unroll
    for (int j = 0; j < 8; ++j)
      acc[j] = __builtin_amdgcn_wmma_f32_16x16x32_bf16(
          false, afrag.v, false, bfrag[j].v, (short)0, acc[j], false, false);
    buf ^= 1;
  }

  // Epilogue: C VGPR r -> M = r + hi*8, N = l16 (ISA 32-bit C/D 16x16 layout)
#pragma unroll
  for (int j = 0; j < 8; ++j) {
#pragma unroll
    for (int r = 0; r < 8; ++r) {
      int row = m0 + wave * 16 + r + hi * 8;
      int col = n0 + j * 16 + l16;
      float v = acc[j][r] + bias[col];
      if (epi == 0) {
        v = v > 0.f ? v : 0.f;
        reinterpret_cast<__bf16*>(Cout)[(size_t)row * N + col] = (__bf16)v;
      } else {
        reinterpret_cast<float*>(Cout)[(size_t)row * N + col] = v;
      }
    }
  }
}

// ---------------------------------------------------------------------------
// block reductions
// ---------------------------------------------------------------------------
__device__ __forceinline__ float block_reduce_sum(float v, float* red) {
  int t = threadIdx.x;
  red[t] = v; __syncthreads();
  for (int off = 128; off > 0; off >>= 1) {
    if (t < off) red[t] += red[t + off];
    __syncthreads();
  }
  float r = red[0]; __syncthreads();
  return r;
}
__device__ __forceinline__ float block_reduce_max(float v, float* red) {
  int t = threadIdx.x;
  red[t] = v; __syncthreads();
  for (int off = 128; off > 0; off >>= 1) {
    if (t < off) red[t] = fmaxf(red[t], red[t + off]);
    __syncthreads();
  }
  float r = red[0]; __syncthreads();
  return r;
}

// ---------------------------------------------------------------------------
// wgbar[d] = mean_s wg_w[d,s] ; wgb_mean = mean(wg_b)
// (slot_imp = hw @ wgbar + wgb_mean  — exact linear identity)
// ---------------------------------------------------------------------------
__global__ void wg_reduce_kernel(const float* __restrict__ wg_w,
                                 const float* __restrict__ wg_b,
                                 float* __restrict__ wgbar,
                                 float* __restrict__ wgb_mean, int D, int S) {
  int d = blockIdx.x * blockDim.x + threadIdx.x;
  if (d < D) {
    float s = 0.f;
    for (int j = 0; j < S; ++j) s += wg_w[(size_t)d * S + j];
    wgbar[d] = s / S;
  }
  if (d == 0) {
    float s = 0.f;
    for (int j = 0; j < S; ++j) s += wg_b[j];
    wgb_mean[0] = s / S;
  }
}

// ---------------------------------------------------------------------------
// x = ff + embed[seq]; LayerNorm -> hiddens (in-place over ff buffer OK);
// slot_imp[bt] = hiddens . wgbar + wgb_mean     (one block per token)
// ---------------------------------------------------------------------------
__global__ __launch_bounds__(256) void residual_ln_kernel(
    const float* __restrict__ ff, const int* __restrict__ seq,
    const float* __restrict__ embed, const float* __restrict__ ln_g,
    const float* __restrict__ ln_b, const float* __restrict__ wgbar,
    const float* __restrict__ wgb_mean, float* __restrict__ hiddens,
    float* __restrict__ slot_imp, int D) {
  __shared__ float red[256];
  const int bt = blockIdx.x;
  const int t = threadIdx.x;
  const float* frow = ff + (size_t)bt * D;
  const float* erow = embed + (size_t)seq[bt] * D;
  float x[4], s = 0.f;
#pragma unroll
  for (int i = 0; i < 4; ++i) {
    int d = t + i * 256;
    x[i] = frow[d] + erow[d];
    s += x[i];
  }
  float mu = block_reduce_sum(s, red) / D;
  float vs = 0.f;
#pragma unroll
  for (int i = 0; i < 4; ++i) { float dl = x[i] - mu; vs += dl * dl; }
  float var = block_reduce_sum(vs, red) / D;
  float rstd = rsqrtf(var + LN_EPS);
  float* hrow = hiddens + (size_t)bt * D;
  float si = 0.f;
#pragma unroll
  for (int i = 0; i < 4; ++i) {
    int d = t + i * 256;
    float hv = (x[i] - mu) * rstd * ln_g[d] + ln_b[d];
    hrow[d] = hv;
    si += hv * wgbar[d];
  }
  float tot = block_reduce_sum(si, red);
  if (t == 0) slot_imp[bt] = tot + wgb_mean[0];
}

// ---------------------------------------------------------------------------
// per-batch top-4 over Tw=T-1, then masked top-2 over first 8 positions
// ---------------------------------------------------------------------------
__global__ void topk_kernel(const float* __restrict__ slot_imp,
                            int* __restrict__ top_idx, int* __restrict__ rev_pos,
                            int T) {
  if (threadIdx.x != 0) return;
  int b = blockIdx.x;
  const float* si = slot_imp + (size_t)b * T;
  int Tw = T - 1;
  int ti[4];
  for (int k = 0; k < 4; ++k) {
    float best = -__builtin_inff(); int bi = 0;
    for (int t = 0; t < Tw; ++t) {
      bool used = false;
      for (int j = 0; j < k; ++j) used |= (ti[j] == t);
      if (!used && si[t] > best) { best = si[t]; bi = t; }
    }
    ti[k] = bi;
    top_idx[b * 4 + k] = bi;
  }
  int rp[2];
  for (int k = 0; k < 2; ++k) {
    float best = -__builtin_inff(); int bi = 0;
    for (int t = 0; t < 8; ++t) {
      bool excl = false;
      for (int j = 0; j < 4; ++j) excl |= (ti[j] == t);
      for (int j = 0; j < k; ++j) excl |= (rp[j] == t);
      if (!excl && si[t] > best) { best = si[t]; bi = t; }
    }
    rp[k] = bi;
    rev_pos[b * 2 + k] = bi;
  }
}

// ---------------------------------------------------------------------------
// mem[b,s,:] : s<4 -> hiddens[top_idx], 4<=s<6 -> hiddens[rev_pos], else memory
// ---------------------------------------------------------------------------
__global__ __launch_bounds__(256) void build_mem_kernel(
    const float* __restrict__ hiddens, const float* __restrict__ memory,
    const int* __restrict__ top_idx, const int* __restrict__ rev_pos,
    float* __restrict__ mem, int T, int D) {
  int b = blockIdx.y, s = blockIdx.x;
  const float* src;
  if (s < 4)      src = hiddens + ((size_t)b * T + top_idx[b * 4 + s]) * D;
  else if (s < 6) src = hiddens + ((size_t)b * T + rev_pos[b * 2 + (s - 4)]) * D;
  else            src = memory + (size_t)s * D;
  float* dst = mem + ((size_t)b * 16 + s) * D;
  for (int d = threadIdx.x; d < D; d += 256) dst[d] = src[d];
}

// ---------------------------------------------------------------------------
// q[b] = query[b] @ rp_w + rp_b   (query = hiddens[b, T-1])
// ---------------------------------------------------------------------------
__global__ __launch_bounds__(256) void rp_gemv_kernel(
    const float* __restrict__ hiddens, const float* __restrict__ rp_w,
    const float* __restrict__ rp_b, float* __restrict__ q, int T, int D) {
  int b = blockIdx.y;
  int dout = blockIdx.x * 256 + threadIdx.x;
  const float* query = hiddens + ((size_t)b * T + (T - 1)) * D;
  float acc = rp_b[dout];
  for (int dp = 0; dp < D; ++dp)
    acc += query[dp] * rp_w[(size_t)dp * D + dout];
  q[(size_t)b * D + dout] = acc;
}

// ---------------------------------------------------------------------------
// L[b,v] = X[b] @ out_w[:,v] + out_b[v] for all 32 b at once.
// One block per 256-column chunk: out_w is read exactly once (HBM-bound).
// ---------------------------------------------------------------------------
__global__ __launch_bounds__(256) void out_gemm_kernel(
    const float* __restrict__ X, long long xstride,
    const float* __restrict__ out_w, const float* __restrict__ out_b,
    float* __restrict__ L, int D, int V) {
  __shared__ float qs[32 * 128];
  int tid = threadIdx.x;
  int v = blockIdx.x * 256 + tid;
  float acc[32];
#pragma unroll
  for (int b = 0; b < 32; ++b) acc[b] = 0.f;
  for (int d0 = 0; d0 < D; d0 += 128) {
    __syncthreads();
#pragma unroll
    for (int i = 0; i < 16; ++i) {
      int idx = tid + i * 256;               // 0..4095 = (b,dd)
      int b = idx >> 7, dd = idx & 127;
      qs[idx] = X[(size_t)b * xstride + d0 + dd];
    }
    __syncthreads();
    if (v < V) {
      for (int d = 0; d < 128; ++d) {
        float w = out_w[(size_t)(d0 + d) * V + v];
#pragma unroll
        for (int b = 0; b < 32; ++b) acc[b] += qs[b * 128 + d] * w;
      }
    }
  }
  if (v < V) {
    float bias = out_b[v];
#pragma unroll
    for (int b = 0; b < 32; ++b) L[(size_t)b * V + v] = acc[b] + bias;
  }
}

// ---------------------------------------------------------------------------
// conf[b] = max softmax = 1 / sum_j exp(l_j - max_j);  use_mem = conf < 0.8
// ---------------------------------------------------------------------------
__global__ __launch_bounds__(256) void conf_kernel(
    const float* __restrict__ L, float* __restrict__ use_mem, int V,
    float thresh) {
  __shared__ float red[256];
  int b = blockIdx.x, t = threadIdx.x;
  const float* row = L + (size_t)b * V;
  float m = -__builtin_inff();
  for (int v = t; v < V; v += 256) m = fmaxf(m, row[v]);
  m = block_reduce_max(m, red);
  float s = 0.f;
  for (int v = t; v < V; v += 256) s += __expf(row[v] - m);
  s = block_reduce_sum(s, red);
  if (t == 0) {
    float conf = 1.f / s;
    use_mem[b] = (conf < thresh) ? 1.f : 0.f;
  }
}

// ---------------------------------------------------------------------------
// scores = mem @ q ; attn = softmax ; ctx = attn @ mem ; blend with query
// ---------------------------------------------------------------------------
__global__ __launch_bounds__(256) void attn_ctx_kernel(
    const float* __restrict__ mem, const float* __restrict__ q,
    const float* __restrict__ hiddens, const float* __restrict__ use_mem,
    float* __restrict__ ctx, int T, int D) {
  __shared__ float red[256];
  __shared__ float sc[16];
  __shared__ float attn[16];
  int b = blockIdx.x, t = threadIdx.x;
  const float* mb = mem + (size_t)b * 16 * D;
  const float* qb = q + (size_t)b * D;
  int s = t >> 4, part = t & 15;
  float p = 0.f;
  for (int i = 0; i < D / 16; ++i) {
    int d = part + i * 16;
    p += mb[(size_t)s * D + d] * qb[d];
  }
  red[t] = p; __syncthreads();
  for (int off = 8; off > 0; off >>= 1) {
    if (part < off) red[t] += red[t + off];
    __syncthreads();
  }
  if (part == 0) sc[s] = red[t];
  __syncthreads();
  if (t == 0) {
    float m = -__builtin_inff();
    for (int i = 0; i < 16; ++i) m = fmaxf(m, sc[i]);
    float sum = 0.f;
    for (int i = 0; i < 16; ++i) { attn[i] = __expf(sc[i] - m); sum += attn[i]; }
    float inv = 1.f / sum;
    for (int i = 0; i < 16; ++i) attn[i] *= inv;
  }
  __syncthreads();
  const float* query = hiddens + ((size_t)b * T + (T - 1)) * D;
  float u = use_mem[b];
  for (int d = t; d < D; d += 256) {
    float c = 0.f;
#pragma unroll
    for (int ss = 0; ss < 16; ++ss) c += attn[ss] * mb[(size_t)ss * D + d];
    ctx[(size_t)b * D + d] = c * u + query[d] * (1.f - u);
  }
}

// ---------------------------------------------------------------------------
// host launcher
// ---------------------------------------------------------------------------
extern "C" void kernel_launch(void* const* d_in, const int* in_sizes, int n_in,
                              void* d_out, int out_size, void* d_ws,
                              size_t ws_size, hipStream_t stream) {
  (void)in_sizes; (void)n_in; (void)out_size; (void)ws_size;
  const int Bv = 32, Tt = 512, Dd = 1024, Vv = 50257, Ss = 16;
  const size_t BT = (size_t)Bv * Tt;

  const int*   seq    = (const int*)d_in[0];
  const float* embed  = (const float*)d_in[1];
  const float* ff_w1  = (const float*)d_in[2];
  const float* ff_b1  = (const float*)d_in[3];
  const float* ff_w2  = (const float*)d_in[4];
  const float* ff_b2  = (const float*)d_in[5];
  const float* ln_g   = (const float*)d_in[6];
  const float* ln_b   = (const float*)d_in[7];
  const float* wg_w   = (const float*)d_in[8];
  const float* wg_b   = (const float*)d_in[9];
  const float* rp_w   = (const float*)d_in[10];
  const float* rp_b   = (const float*)d_in[11];
  const float* out_w  = (const float*)d_in[12];
  const float* out_b  = (const float*)d_in[13];
  const float* memory = (const float*)d_in[14];
  float* out = (float*)d_out;

  char* ws = (char*)d_ws;
  size_t off = 0;
  auto alloc = [&](size_t bytes) -> char* {
    char* p = ws + off;
    off += (bytes + 255) & ~(size_t)255;
    return p;
  };

  __bf16* w1b  = (__bf16*)alloc((size_t)Dd * 2 * Dd * 2);
  __bf16* w2b  = (__bf16*)alloc((size_t)2 * Dd * Dd * 2);
  __bf16* hbf  = (__bf16*)alloc(BT * Dd * 2);
  __bf16* act1 = (__bf16*)alloc(BT * 2 * Dd * 2);
  float*  ffo  = (float*)alloc(BT * Dd * 4);        // also reused as hiddens
  float*  hid  = ffo;                               // LN is in-place safe
  float*  simp = (float*)alloc((size_t)Bv * Tt * 4);
  float*  wgbar = (float*)alloc((size_t)Dd * 4);
  float*  wgbm  = (float*)alloc(256);
  int*    tidx  = (int*)alloc((size_t)Bv * 4 * 4);
  int*    rpos  = (int*)alloc((size_t)Bv * 2 * 4);
  float*  memb  = (float*)alloc((size_t)Bv * Ss * Dd * 4);
  float*  qb    = (float*)alloc((size_t)Bv * Dd * 4);
  float*  Ld    = (float*)alloc((size_t)Bv * Vv * 4);
  float*  usem  = (float*)alloc(256);
  float*  ctx   = (float*)alloc((size_t)Bv * Dd * 4);

  // 1. weight conversion + wg reduction
  f32_to_bf16_kernel<<<(Dd * 2 * Dd / 4 + 255) / 256, 256, 0, stream>>>(
      ff_w1, w1b, (size_t)Dd * 2 * Dd / 4);
  f32_to_bf16_kernel<<<(2 * Dd * Dd / 4 + 255) / 256, 256, 0, stream>>>(
      ff_w2, w2b, (size_t)2 * Dd * Dd / 4);
  wg_reduce_kernel<<<(Dd + 255) / 256, 256, 0, stream>>>(wg_w, wg_b, wgbar,
                                                         wgbm, Dd, Ss);
  // 2. embedding gather
  gather_embed_bf16_kernel<<<(unsigned)(BT * Dd / 4 / 256), 256, 0, stream>>>(
      seq, embed, hbf, Dd);
  // 3. FFN via WMMA (+TDM staging)
  wmma_ffn_gemm<<<dim3(2 * Dd / 128, (unsigned)(BT / 128)), 256, 0, stream>>>(
      hbf, w1b, ff_b1, act1, 2 * Dd, Dd, /*epi=*/0);
  wmma_ffn_gemm<<<dim3(Dd / 128, (unsigned)(BT / 128)), 256, 0, stream>>>(
      act1, w2b, ff_b2, ffo, Dd, 2 * Dd, /*epi=*/1);
  // 4. residual + LN + slot_imp
  residual_ln_kernel<<<(unsigned)BT, 256, 0, stream>>>(
      ffo, seq, embed, ln_g, ln_b, wgbar, wgbm, hid, simp, Dd);
  // 5. selection
  topk_kernel<<<Bv, 32, 0, stream>>>(simp, tidx, rpos, Tt);
  // 6. direct logits + confidence
  out_gemm_kernel<<<(Vv + 255) / 256, 256, 0, stream>>>(
      hid + (size_t)(Tt - 1) * Dd, (long long)Tt * Dd, out_w, out_b, Ld, Dd, Vv);
  conf_kernel<<<Bv, 256, 0, stream>>>(Ld, usem, Vv, 0.8f);
  // 7. memory assembly + attention
  build_mem_kernel<<<dim3(Ss, Bv), 256, 0, stream>>>(hid, memory, tidx, rpos,
                                                     memb, Tt, Dd);
  rp_gemv_kernel<<<dim3(Dd / 256, Bv), 256, 0, stream>>>(hid, rp_w, rp_b, qb,
                                                         Tt, Dd);
  attn_ctx_kernel<<<Bv, 256, 0, stream>>>(memb, qb, hid, usem, ctx, Tt, Dd);
  // 8. final logits
  out_gemm_kernel<<<(Vv + 255) / 256, 256, 0, stream>>>(
      ctx, (long long)Dd, out_w, out_b, out, Dd, Vv);
}
